// AxialAttention_64742337020025
// MI455X (gfx1250) — matrix operand order repaired and checked by reading, compiled, not run on hardware
//
#include <hip/hip_runtime.h>

// ---------- problem constants ----------
constexpr int kB = 2;
constexpr int kC = 768;
constexpr int kS = 13824;        // 24*24*24
constexpr int kHeads = 12;
constexpr int kL = 24;           // axial sequence length

typedef __attribute__((ext_vector_type(16))) _Float16 v16h;
typedef __attribute__((ext_vector_type(8)))  _Float16 v8h;
typedef __attribute__((ext_vector_type(4)))  _Float16 v4h;
typedef __attribute__((ext_vector_type(2)))  _Float16 v2h;
typedef __attribute__((ext_vector_type(8)))  float    v8f;
typedef __attribute__((ext_vector_type(4)))  float    v4f;

union V16u { v16h v; v8h h[2]; };

__device__ inline v16h frag16(const _Float16* p0, const _Float16* p1) {
    V16u u;
    u.h[0] = *(const v8h*)p0;
    u.h[1] = *(const v8h*)p1;
    return u.v;
}

__device__ inline v8f wmma_f16(v16h a, v16h b, v8f c) {
    // D = A(16x32 f16) * B(32x16 f16) + C(f32)
    return __builtin_amdgcn_wmma_f32_16x16x32_f16(false, a, false, b,
                                                  (short)0, c, false, false);
}

// ---- CDNA5 async global->LDS copy (ASYNCcnt path), 32 bytes per lane ----
__device__ inline void async_copy32(const _Float16* gp, _Float16* lp) {
    unsigned lds = (unsigned)(unsigned long long)(void*)lp;      // LDS byte offset
    unsigned long long g = (unsigned long long)(const void*)gp;  // global address
    asm volatile("global_load_async_to_lds_b128 %0, %1, off\n\t"
                 "global_load_async_to_lds_b128 %0, %1, off offset:16"
                 :: "v"(lds), "v"(g) : "memory");
}
__device__ inline void wait_async0() {
    asm volatile("s_wait_asynccnt 0" ::: "memory");
}

// ---------------------------------------------------------------------------
// Kernel 0: one-time fp32 -> f16 weight conversion
// ---------------------------------------------------------------------------
__global__ __launch_bounds__(256)
void k_cvt_w(const float* __restrict__ w, _Float16* __restrict__ wh, int n4) {
    int i = blockIdx.x * 256 + threadIdx.x;
    if (i < n4) {
        v4f v = *(const v4f*)(w + (size_t)i * 4);
        v4h o;
        o[0] = (_Float16)v.x; o[1] = (_Float16)v.y;
        o[2] = (_Float16)v.z; o[3] = (_Float16)v.w;
        *(v4h*)(wh + (size_t)i * 4) = o;
    }
}

// ---------------------------------------------------------------------------
// Kernel 1: InstanceNorm over spatial dims per (b,c); fp32 in, f16 out [b][c][s]
// ---------------------------------------------------------------------------
__global__ __launch_bounds__(256)
void k_instnorm1(const float* __restrict__ x, _Float16* __restrict__ xn) {
    __shared__ float ssum[256], ssq[256];
    const int t = threadIdx.x;
    const size_t base = (size_t)blockIdx.x * kS;      // block = b*768 + c
    const v4f* xv = (const v4f*)(x + base);
    float sum = 0.f, sq = 0.f;
    for (int i = t; i < kS / 4; i += 256) {
        v4f v = xv[i];
        sum += v.x + v.y + v.z + v.w;
        sq  += v.x * v.x + v.y * v.y + v.z * v.z + v.w * v.w;
    }
    ssum[t] = sum; ssq[t] = sq;
    __syncthreads();
    for (int off = 128; off > 0; off >>= 1) {
        if (t < off) { ssum[t] += ssum[t + off]; ssq[t] += ssq[t + off]; }
        __syncthreads();
    }
    const float mean = ssum[0] * (1.f / kS);
    const float var  = ssq[0] * (1.f / kS) - mean * mean;
    const float rs   = rsqrtf(var + 1e-5f);
    for (int i = t; i < kS / 4; i += 256) {
        v4f v = xv[i];
        v4h o;
        o[0] = (_Float16)((v.x - mean) * rs);
        o[1] = (_Float16)((v.y - mean) * rs);
        o[2] = (_Float16)((v.z - mean) * rs);
        o[3] = (_Float16)((v.w - mean) * rs);
        *(v4h*)(xn + base + (size_t)i * 4) = o;
    }
}

// ---------------------------------------------------------------------------
// Tiled WMMA GEMM:  Y[o][s] = sum_c Wh[o][c] * Bm[c][s]  (+bias)
// 128x128 output tile per 256-thread block, K-step 32, double-buffered LDS.
// A tile staged with CDNA5 async global->LDS DMA (ASYNCcnt); B tile staged
// through VGPRs with a packed-pair transpose.  8 waves (2x4), each wave owns
// a 64x32 region = 4x2 wmma accumulators.
// MODE 0: write f16 into qkv head layout [b][3][heads][s][64] (LDS restage)
// MODE 1: write f32 [b][o][s] directly (final output)
// ---------------------------------------------------------------------------
template<int MODE>
__global__ __launch_bounds__(256)
void k_gemm(const _Float16* __restrict__ Wh, const float* __restrict__ bias,
            const _Float16* __restrict__ Bm, void* __restrict__ outp, int O) {
    // two stages; each stage: As [128][40] at +0, Bt [128][40] at +5120
    __shared__ _Float16 smem[2][10240];               // 40 KB static LDS

    const int t    = threadIdx.x;
    const int lane = t & 31;
    const int wv   = t >> 5;
    const int wm   = (wv & 1) * 64;                   // wave M offset
    const int wn   = (wv >> 1) * 32;                  // wave N offset
    const int o0   = blockIdx.y * 128;
    const int s0   = blockIdx.x * 128;
    const int b    = blockIdx.z;

    const _Float16* Bb = Bm + (size_t)b * kC * kS;

    v8f acc[4][2] = {};

    // A loader: thread t -> row am (0..127), half akh (0/16); 32 bytes each
    const int am  = t >> 1, akh = (t & 1) * 16;
    // B loader: thread t -> k-pair kp (0,2,..,30), n-group nh (0,8,..,120)
    const int kp  = (t & 15) * 2, nh = (t >> 4) * 8;

    const int kbA  = (lane < 16) ? 0 : 8;
    const int kbB  = (lane < 16) ? 0 : 8;
    const int arow = lane & 15;

    const _Float16* wrow = Wh + (size_t)(o0 + am) * kC + akh;

    v8h b0, b1;                                       // B prefetch registers
    auto gloadB = [&](int kt) {
        const _Float16* bp = Bb + (size_t)(kt + kp) * kS + s0 + nh;
        b0 = *(const v8h*)bp;                          // c-row kp
        b1 = *(const v8h*)(bp + kS);                   // c-row kp+1
    };
    auto sstoreB = [&](int st) {
        _Float16* Bt = smem[st] + 5120;
#pragma unroll
        for (int i = 0; i < 8; i++) {                  // packed k-pair (b32) stores
            v2h p; p[0] = b0[i]; p[1] = b1[i];
            *(v2h*)(Bt + (nh + i) * 40 + kp) = p;
        }
    };

    const int NT = kC / 32;                            // 24 K-tiles
    async_copy32(wrow, smem[0] + am * 40 + akh);       // A tile 0 via DMA
    gloadB(0);
    sstoreB(0);
    wait_async0();
    __syncthreads();

    for (int kt = 0; kt < NT; ++kt) {
        const int cur = kt & 1;
        if (kt + 1 < NT) {
            async_copy32(wrow + (kt + 1) * 32,         // next A tile via DMA
                         smem[cur ^ 1] + am * 40 + akh);
            gloadB((kt + 1) * 32);                     // next B tile into regs
        }
        if (kt + 2 < NT)                               // deep B prefetch into L2/L0
            __builtin_prefetch(Bb + (size_t)((kt + 2) * 32 + kp) * kS + s0 + nh, 0, 0);

        const _Float16* As = smem[cur];
        const _Float16* Bt = smem[cur] + 5120;
        v16h bf[2];
#pragma unroll
        for (int nt = 0; nt < 2; ++nt) {
            const _Float16* r = Bt + (wn + nt * 16 + arow) * 40 + kbB;
            bf[nt] = frag16(r, r + 16);
        }
#pragma unroll
        for (int mt = 0; mt < 4; ++mt) {
            const _Float16* r = As + (wm + mt * 16 + arow) * 40;
            v16h af = frag16(r + kbA, r + 16 + kbA);
#pragma unroll
            for (int nt = 0; nt < 2; ++nt)
                acc[mt][nt] = wmma_f16(af, bf[nt], acc[mt][nt]);
        }

        if (kt + 1 < NT) sstoreB(cur ^ 1);             // fill other stage
        wait_async0();                                 // A DMA for next stage done
        __syncthreads();
    }

    const int mrow = (lane >> 4) * 8;
    const int ncol = lane & 15;

    if (MODE == 1) {
        float* out = (float*)outp;
#pragma unroll
        for (int mt = 0; mt < 4; ++mt)
#pragma unroll
            for (int nt = 0; nt < 2; ++nt)
#pragma unroll
                for (int j = 0; j < 8; ++j) {
                    int m = wm + mt * 16 + j + mrow;
                    int n = wn + nt * 16 + ncol;
                    out[((size_t)b * O + o0 + m) * kS + s0 + n] =
                        acc[mt][nt][j] + bias[o0 + m];
                }
    } else {
        _Float16* D = &smem[0][0];                     // [128][136] f16 (34 KB)
#pragma unroll
        for (int mt = 0; mt < 4; ++mt)
#pragma unroll
            for (int nt = 0; nt < 2; ++nt)
#pragma unroll
                for (int j = 0; j < 8; ++j) {
                    int m = wm + mt * 16 + j + mrow;
                    int n = wn + nt * 16 + ncol;
                    D[m * 136 + n] = (_Float16)(acc[mt][nt][j] + bias[o0 + m]);
                }
        __syncthreads();
        _Float16* out = (_Float16*)outp;
        for (int i = t; i < 128 * 128; i += 256) {
            int ol = i & 127, sl = i >> 7;
            int o = o0 + ol;
            int wh = o / kC, rem = o - wh * kC;
            int he = rem >> 6, hd = rem & 63;
            out[((((size_t)b * 3 + wh) * kHeads + he) * kS + s0 + sl) * 64 + hd] =
                D[ol * 136 + sl];
        }
    }
}

// ---------------------------------------------------------------------------
// Kernel 3: RMSNorm over head dim (64) for q and k, in place on f16 qkv
// ---------------------------------------------------------------------------
__global__ __launch_bounds__(256)
void k_rmsnorm_qk(_Float16* __restrict__ qkv, const float* __restrict__ qs,
                  const float* __restrict__ ks) {
    __shared__ float sc[2][64];
    const int t = threadIdx.x;
    if (t < 64)       sc[0][t] = qs[t];
    else if (t < 128) sc[1][t - 64] = ks[t - 64];
    __syncthreads();

    const int gid = blockIdx.x * 256 + t;          // 0 .. 663551 exactly
    const int per_b = 2 * kHeads * kS;
    const int b  = gid / per_b;
    const int r  = gid - b * per_b;
    const int wh = r / (kHeads * kS);
    const int r2 = r - wh * (kHeads * kS);

    _Float16* p = qkv + ((size_t)(b * 3 + wh) * kHeads * kS + r2) * 64;
    v8h h[8];
    float sq = 0.f;
#pragma unroll
    for (int i = 0; i < 8; i++) {
        h[i] = *(const v8h*)(p + i * 8);
#pragma unroll
        for (int j = 0; j < 8; j++) { float f = (float)h[i][j]; sq += f * f; }
    }
    const float rs = rsqrtf(sq * (1.f / 64.f) + 1e-6f);
#pragma unroll
    for (int i = 0; i < 8; i++) {
#pragma unroll
        for (int j = 0; j < 8; j++)
            h[i][j] = (_Float16)((float)h[i][j] * sc[wh][i * 8 + j] * rs);
        *(v8h*)(p + i * 8) = h[i];
    }
}

// ---------------------------------------------------------------------------
// Axial attention: one wave per row (L=24, hd=64), 2 waves per block.
// Logits 24x24 via 8 wmma (padded to 32), softmax in LDS, P*V via 8 wmma.
// ---------------------------------------------------------------------------
struct WaveLds {
    _Float16 Qs[32 * 72];   // [m][hd], row stride 72 (144B, 16B aligned)
    _Float16 Ks[32 * 72];   // [n][hd]
    _Float16 Vt[64 * 40];   // [hd][pos], row stride 40
    _Float16 Ps[32 * 40];   // softmax probs [m][keypos]
    float    Ls[32 * 33];   // logits f32
};

template<int AXIS, bool FIRST>
__global__ __launch_bounds__(64)
void k_attn(const _Float16* __restrict__ qkv, float* __restrict__ y) {
    __shared__ WaveLds wls[2];
    const int t = threadIdx.x;
    const int lane = t & 31;
    const int wv = t >> 5;
    WaveLds& w = wls[wv];

    const int P  = blockIdx.x * 2 + wv;           // 0 .. 13823
    const int b  = P / (kHeads * 576);
    const int r  = P - b * (kHeads * 576);
    const int he = r / 576;
    const int uv = r - he * 576;

    int base_s, stride;
    if (AXIS == 0)      { base_s = uv; stride = 576; }                         // fixed (w,d)
    else if (AXIS == 1) { base_s = (uv / 24) * 576 + (uv % 24); stride = 24; } // fixed (h,d)
    else                { base_s = (uv / 24) * 576 + (uv % 24) * 24; stride = 1; } // fixed (h,w)

    const size_t headvol = (size_t)kHeads * kS * 64;
    const size_t qoff = (((size_t)b * 3 + 0) * kHeads + he) * (size_t)kS * 64 +
                        (size_t)base_s * 64;
    const size_t koff = qoff + headvol;
    const size_t voff = koff + headvol;
    const size_t st64 = (size_t)stride * 64;

    // ---- stage Q,K,V into LDS; zero all padding ----
    if (lane < kL) {
        const _Float16* qp = qkv + qoff + lane * st64;
        const _Float16* kp = qkv + koff + lane * st64;
        const _Float16* vp = qkv + voff + lane * st64;
#pragma unroll
        for (int i = 0; i < 8; i++) {
            *(v8h*)(w.Qs + lane * 72 + i * 8) = *(const v8h*)(qp + i * 8);
            *(v8h*)(w.Ks + lane * 72 + i * 8) = *(const v8h*)(kp + i * 8);
        }
#pragma unroll
        for (int i = 0; i < 8; i++) {
            v8h v = *(const v8h*)(vp + i * 8);
#pragma unroll
            for (int j = 0; j < 8; j++) w.Vt[(i * 8 + j) * 40 + lane] = v[j];
        }
    } else {
        v8h z = {};
#pragma unroll
        for (int i = 0; i < 8; i++) {
            *(v8h*)(w.Qs + lane * 72 + i * 8) = z;
            *(v8h*)(w.Ks + lane * 72 + i * 8) = z;
        }
#pragma unroll
        for (int i = 0; i < 4; i++) *(v8h*)(w.Ps + lane * 40 + i * 8) = z;
    }
    {   // zero V padding columns 24..31 for all hd rows
        v8h z = {};
        *(v8h*)(w.Vt + lane * 40 + 24)        = z;
        *(v8h*)(w.Vt + (lane + 32) * 40 + 24) = z;
    }
    __syncthreads();

    const int kbA  = (lane < 16) ? 0 : 8;
    const int kbB  = (lane < 16) ? 0 : 8;
    const int arow = lane & 15;
    const int mrow = (lane >> 4) * 8;

    // ---- logits = Q * K^T * (1/sqrt(64)) ----
#pragma unroll
    for (int mt = 0; mt < 2; ++mt)
#pragma unroll
        for (int nt = 0; nt < 2; ++nt) {
            v8f acc = {};
#pragma unroll
            for (int kk = 0; kk < 2; ++kk) {
                const _Float16* qr = w.Qs + (mt * 16 + arow) * 72 + kk * 32;
                v16h a = frag16(qr + kbA, qr + 16 + kbA);
                const _Float16* kr = w.Ks + (nt * 16 + arow) * 72 + kk * 32 + kbB;
                v16h bb = frag16(kr, kr + 16);
                acc = wmma_f16(a, bb, acc);
            }
#pragma unroll
            for (int j = 0; j < 8; ++j)
                w.Ls[(mt * 16 + j + mrow) * 33 + nt * 16 + arow] = acc[j] * 0.125f;
        }
    __syncthreads();

    // ---- softmax per query row (lanes 0..23) ----
    if (lane < kL) {
        const float* row = w.Ls + lane * 33;
        float mx = -1e30f;
#pragma unroll
        for (int j = 0; j < kL; j++) mx = fmaxf(mx, row[j]);
        float e[kL]; float s = 0.f;
#pragma unroll
        for (int j = 0; j < kL; j++) { e[j] = __expf(row[j] - mx); s += e[j]; }
        const float inv = 1.f / s;
#pragma unroll
        for (int j = 0; j < kL; j++) w.Ps[lane * 40 + j] = (_Float16)(e[j] * inv);
#pragma unroll
        for (int j = kL; j < 32; j++) w.Ps[lane * 40 + j] = (_Float16)0.f;
    }
    __syncthreads();

    // ---- Y = P * V ; accumulate (sum of 3 axes)/3 into y ----
    const size_t ybase = (((size_t)b * kHeads + he) * kS + base_s) * 64;
#pragma unroll
    for (int mt = 0; mt < 2; ++mt) {
        const _Float16* pr = w.Ps + (mt * 16 + arow) * 40;
        v16h a = frag16(pr + kbA, pr + 16 + kbA);
#pragma unroll
        for (int nt = 0; nt < 4; ++nt) {
            const _Float16* vr = w.Vt + (nt * 16 + arow) * 40 + kbB;
            v16h bb = frag16(vr, vr + 16);
            v8f acc = {};
            acc = wmma_f16(a, bb, acc);
#pragma unroll
            for (int j = 0; j < 8; ++j) {
                int m = mt * 16 + j + mrow;
                if (m < kL) {
                    size_t addr = ybase + (size_t)m * st64 + nt * 16 + arow;
                    float val = acc[j] * (1.f / 3.f);
                    if (FIRST) y[addr] = val; else y[addr] += val;
                }
            }
        }
    }
}

// ---------------------------------------------------------------------------
// InstanceNorm #2 over y [b][he][s][64]: stats then apply + transpose to
// f16 xn2 [b][c][s] for the proj GEMM.
// ---------------------------------------------------------------------------
__global__ __launch_bounds__(256)
void k_in2_stats(const float* __restrict__ y, float* __restrict__ mean2,
                 float* __restrict__ rsig2) {
    __shared__ float ss[256], sq[256];
    const int t = threadIdx.x;
    const int hd = t & 63, g = t >> 6;
    const float* base = y + (size_t)blockIdx.x * kS * 64;   // block = b*12 + he
    float s = 0.f, q = 0.f;
    for (int sp = g; sp < kS; sp += 4) {
        float v = base[(size_t)sp * 64 + hd];
        s += v; q += v * v;
    }
    ss[t] = s; sq[t] = q;
    __syncthreads();
    if (t < 64) {
        float S1 = ss[t] + ss[t + 64] + ss[t + 128] + ss[t + 192];
        float Q1 = sq[t] + sq[t + 64] + sq[t + 128] + sq[t + 192];
        float mean = S1 * (1.f / kS);
        float var  = Q1 * (1.f / kS) - mean * mean;
        int b = blockIdx.x / kHeads, he = blockIdx.x % kHeads;
        mean2[b * kC + he * 64 + hd] = mean;
        rsig2[b * kC + he * 64 + hd] = rsqrtf(var + 1e-5f);
    }
}

__global__ __launch_bounds__(256)
void k_in2_apply(const float* __restrict__ y, const float* __restrict__ mean2,
                 const float* __restrict__ rsig2, _Float16* __restrict__ xn2) {
    __shared__ _Float16 T[64 * 68];
    __shared__ float m_s[64], r_s[64];
    const int t = threadIdx.x;
    const int blk = blockIdx.x;                 // b * (12*216) + he*216 + sblk
    const int b  = blk / (kHeads * 216);
    const int r  = blk - b * (kHeads * 216);
    const int he = r / 216;
    const int s0 = (r - he * 216) * 64;

    if (t < 64) {
        m_s[t] = mean2[b * kC + he * 64 + t];
        r_s[t] = rsig2[b * kC + he * 64 + t];
    }
    __syncthreads();

    const float* yb = y + (((size_t)b * kHeads + he) * kS + s0) * 64;
#pragma unroll
    for (int i = 0; i < 16; i++) {
        int idx = i * 256 + t;
        int sl = idx >> 6, hd = idx & 63;
        float v = yb[(size_t)sl * 64 + hd];
        T[hd * 68 + sl] = (_Float16)((v - m_s[hd]) * r_s[hd]);
    }
    __syncthreads();
    _Float16* ob = xn2 + ((size_t)b * kC + he * 64) * kS + s0;
#pragma unroll
    for (int i = 0; i < 16; i++) {
        int idx = i * 256 + t;
        int hd = idx >> 6, sl = idx & 63;
        ob[(size_t)hd * kS + sl] = T[hd * 68 + sl];
    }
}

// ---------------------------------------------------------------------------
// Host: pipeline launch
// ---------------------------------------------------------------------------
extern "C" void kernel_launch(void* const* d_in, const int* in_sizes, int n_in,
                              void* d_out, int out_size, void* d_ws, size_t ws_size,
                              hipStream_t stream) {
    (void)in_sizes; (void)n_in; (void)out_size; (void)ws_size;
    const float* x       = (const float*)d_in[0];
    const float* w_qkv   = (const float*)d_in[1];
    const float* b_qkv   = (const float*)d_in[2];
    const float* q_scale = (const float*)d_in[3];
    const float* k_scale = (const float*)d_in[4];
    const float* w_proj  = (const float*)d_in[5];
    const float* b_proj  = (const float*)d_in[6];
    float* out = (float*)d_out;

    char* ws = (char*)d_ws;
    size_t off = 0;
    auto carve = [&](size_t bytes) {
        void* p = ws + off;
        off += (bytes + 255) & ~(size_t)255;
        return p;
    };
    _Float16* xn    = (_Float16*)carve((size_t)kB * kC * kS * 2);       // 42.5 MB
    _Float16* qkv   = (_Float16*)carve((size_t)kB * 3 * kC * kS * 2);   // 127 MB
    float*    yb    = (float*)   carve((size_t)kB * kC * kS * 4);       // 85 MB
    _Float16* xn2   = (_Float16*)carve((size_t)kB * kC * kS * 2);       // 42.5 MB
    _Float16* wqkvh = (_Float16*)carve((size_t)3 * kC * kC * 2);        // 3.5 MB
    _Float16* wprojh= (_Float16*)carve((size_t)kC * kC * 2);            // 1.2 MB
    float*    mean2 = (float*)   carve(kB * kC * 4);
    float*    rsig2 = (float*)   carve(kB * kC * 4);

    // 0. one-time weight conversion fp32 -> f16
    k_cvt_w<<<(3 * kC * kC / 4 + 255) / 256, 256, 0, stream>>>(w_qkv, wqkvh,
                                                               3 * kC * kC / 4);
    k_cvt_w<<<(kC * kC / 4 + 255) / 256, 256, 0, stream>>>(w_proj, wprojh,
                                                           kC * kC / 4);
    // 1. InstanceNorm(x) -> f16
    k_instnorm1<<<kB * kC, 256, 0, stream>>>(x, xn);
    // 2. QKV GEMM (2304x768 x 768x13824 per batch), head-layout f16 output
    k_gemm<0><<<dim3(kS / 128, 2304 / 128, kB), 256, 0, stream>>>(
        wqkvh, b_qkv, xn, (void*)qkv, 2304);
    // 3. RMSNorm q,k
    k_rmsnorm_qk<<<(kB * 2 * kHeads * kS) / 256, 256, 0, stream>>>(
        qkv, q_scale, k_scale);
    // 4-6. Axial attention along h, w, d (accumulated /3 into y)
    k_attn<0, true ><<<(kB * kHeads * 576) / 2, 64, 0, stream>>>(qkv, yb);
    k_attn<1, false><<<(kB * kHeads * 576) / 2, 64, 0, stream>>>(qkv, yb);
    k_attn<2, false><<<(kB * kHeads * 576) / 2, 64, 0, stream>>>(qkv, yb);
    // 7-8. InstanceNorm(y) -> f16 [b][c][s]
    k_in2_stats<<<kB * kHeads, 256, 0, stream>>>(yb, mean2, rsig2);
    k_in2_apply<<<kB * kHeads * 216, 256, 0, stream>>>(yb, mean2, rsig2, xn2);
    // 9. proj GEMM -> fp32 output
    k_gemm<1><<<dim3(kS / 128, kC / 128, kB), 256, 0, stream>>>(
        wprojh, b_proj, xn2, (void*)out, kC);
}
